// GAT_652835029800
// MI455X (gfx1250) — matrix operand order repaired and checked by reading, not compiled, not run on hardware
//
#include <hip/hip_runtime.h>
#include <hip/hip_bf16.h>

// ---------------------------------------------------------------------------
// GAT (2-layer, PyG GATConv eval mode) for MI455X / gfx1250.
// GEMMs via v_wmma_f32_16x16x32_f16: one wave owns a 16xNcol strip, reusing
// the A fragment across NT column tiles (fully unrolled -> 32/16 WMMAs/wave).
// Edge softmax-aggregation: wave-per-edge, f32 atomics (L2-resident).
// ---------------------------------------------------------------------------

typedef __attribute__((ext_vector_type(16))) _Float16 v16h;
typedef __attribute__((ext_vector_type(8)))  float    v8f;

union AV  { v16h h; unsigned int u[8]; };
union P2  { unsigned int u; _Float16 h[2]; };

#define NEG_SLOPE 0.2f
#define HEADS 4
#define HID   32
#define INF_  128   // IN = HEADS*HID = 128
#define OUTF  64
#define KDIM  128   // K of both GEMMs

// ---------------- utility kernels -----------------------------------------

__global__ void gat_zero_f32(float* __restrict__ p, int n) {
  int i = blockIdx.x * blockDim.x + threadIdx.x;
  int s = gridDim.x * blockDim.x;
  for (; i < n; i += s) p[i] = 0.0f;
}

__global__ void gat_cvt_f16(const float* __restrict__ x, _Float16* __restrict__ y, int n) {
  int i = blockIdx.x * blockDim.x + threadIdx.x;
  int s = gridDim.x * blockDim.x;
  for (; i < n; i += s) y[i] = (_Float16)x[i];
}

// Pack row-major W[K,Ncol] (f32) into Wp[kp*Ncol + n] = (f16(W[2kp][n]), f16(W[2kp+1][n]))
// so each WMMA B-operand VGPR (K pair per dword, N across lanes) is one dword load.
__global__ void gat_pack_w(const float* __restrict__ W, unsigned int* __restrict__ Wp,
                           int K2, int Ncol) {
  int i = blockIdx.x * blockDim.x + threadIdx.x;
  if (i >= K2 * Ncol) return;
  int kp = i / Ncol, n = i - kp * Ncol;
  P2 p;
  p.h[0] = (_Float16)W[(2 * kp)     * Ncol + n];
  p.h[1] = (_Float16)W[(2 * kp + 1) * Ncol + n];
  Wp[kp * Ncol + n] = p.u;
}

// ---------------- WMMA GEMM: C[M, NT*16] = A[M,128] * B[128, NT*16] --------
// A32 : f16 row-major, read as packed dwords [M, 64]; two b128 loads/k-step.
// Bp32: packed (k,k+1) pairs from gat_pack_w.
// One wave computes a 16 x (NT*16) strip; A fragment reused across NT tiles.
template <int NT>
__global__ __launch_bounds__(128)
void gat_gemm_wmma(const unsigned int* __restrict__ A32,
                   const unsigned int* __restrict__ Bp32,
                   float* __restrict__ C, int M) {
  const int Ncol = NT * 16;
  const int lane = threadIdx.x & 31;
  const int wave = blockIdx.x * (blockDim.x >> 5) + (threadIdx.x >> 5);
  if (wave * 16 >= M) return;               // wave-uniform (EXEC stays all-1)

  const int half = lane >> 4;               // 0: lanes 0-15, 1: lanes 16-31
  const int mr   = lane & 15;
  const int row  = wave * 16 + mr;
  const int K2   = KDIM >> 1;

  v8f acc[NT];
#pragma unroll
  for (int j = 0; j < NT; ++j) acc[j] = {};

#pragma unroll
  for (int kk = 0; kk < KDIM; kk += 32) {
    // A 16x32 f16 fragment (ISA 7.12.2): u[0..3] -> k = kk+8*half+{0,2,4,6},
    // u[4..7] -> +16. Dword-contiguous -> two uint4 (b128) loads.
    AV a;
    const uint4* ap =
        reinterpret_cast<const uint4*>(A32 + row * K2 + (kk >> 1) + half * 4);
    uint4 a0 = ap[0];
    uint4 a1 = ap[2];                        // +8 dwords
    a.u[0] = a0.x; a.u[1] = a0.y; a.u[2] = a0.z; a.u[3] = a0.w;
    a.u[4] = a1.x; a.u[5] = a1.y; a.u[6] = a1.z; a.u[7] = a1.w;

#pragma unroll
    for (int j = 0; j < NT; ++j) {
      AV b;                                  // B 32x16: VGPR v -> K pair (8*half+2(v&3)+16*(v>>2))
      const int nc = j * 16 + (lane & 15);
#pragma unroll
      for (int v = 0; v < 8; ++v) {
        int kp = (kk >> 1) + (half << 2) + (v & 3) + ((v >> 2) << 3);   // row of Bp32
        b.u[v] = Bp32[kp * Ncol + nc];
      }
      acc[j] = __builtin_amdgcn_wmma_f32_16x16x32_f16(false, a.h, false, b.h,
                                                      (short)0, acc[j],
                                                      false, false);
    }
  }

#pragma unroll
  for (int j = 0; j < NT; ++j) {
#pragma unroll
    for (int r = 0; r < 8; ++r) {            // C layout: VGPR r -> M = 8*half + r
      int m = wave * 16 + (half << 3) + r;
      C[m * Ncol + j * 16 + (lane & 15)] = acc[j][r];
    }
  }
}

// ---------------- layer 1: attention coefficients --------------------------

__global__ void gat_att1(const float* __restrict__ H1, const float* __restrict__ as,
                         const float* __restrict__ ad, float* __restrict__ a_s,
                         float* __restrict__ a_d, int N) {
  int i = blockIdx.x * blockDim.x + threadIdx.x;   // node*HEADS + head
  if (i >= N * HEADS) return;
  int node = i >> 2, head = i & 3;
  const float* h = H1 + node * INF_ + head * HID;
  float s = 0.f, d = 0.f;
#pragma unroll
  for (int c = 0; c < HID; ++c) {
    float v = h[c];
    s += v * as[head * HID + c];
    d += v * ad[head * HID + c];
  }
  a_s[i] = s; a_d[i] = d;
}

__global__ void gat_edge1(const int* __restrict__ src, const int* __restrict__ dst,
                          const float* __restrict__ a_s, const float* __restrict__ a_d,
                          float* __restrict__ ew, float* __restrict__ denom, int E) {
  int e = blockIdx.x * blockDim.x + threadIdx.x;
  if (e >= E) return;
  int s = src[e], d = dst[e];
#pragma unroll
  for (int h = 0; h < HEADS; ++h) {
    float v = a_s[s * HEADS + h] + a_d[d * HEADS + h];
    v = v > 0.f ? v : NEG_SLOPE * v;                 // leaky relu
    float w = expf(v);                               // softmax w/o max-shift (shift-invariant)
    ew[e * HEADS + h] = w;
    atomicAdd(&denom[d * HEADS + h], w);
  }
}

// wave-per-edge: lane l covers channels 4l..4l+3 (head = l>>3)
__global__ __launch_bounds__(256)
void gat_agg1(const int* __restrict__ src, const int* __restrict__ dst,
              const float* __restrict__ H1, const float* __restrict__ ew,
              const float* __restrict__ denom, float* __restrict__ agg, int E) {
  int g = blockIdx.x * blockDim.x + threadIdx.x;
  int e = g >> 5, l = g & 31;
  if (e >= E) return;
  int s = src[e], d = dst[e];
  int head = l >> 3;
  float coeff = ew[e * HEADS + head] / (denom[d * HEADS + head] + 1e-16f);
  const float4 hv = *reinterpret_cast<const float4*>(H1 + s * INF_ + l * 4);
  float* o = agg + d * INF_ + l * 4;
  atomicAdd(o + 0, coeff * hv.x);
  atomicAdd(o + 1, coeff * hv.y);
  atomicAdd(o + 2, coeff * hv.z);
  atomicAdd(o + 3, coeff * hv.w);
}

// bias + ReLU + convert to f16 for GEMM2
__global__ void gat_post1(const float* __restrict__ agg, const float* __restrict__ b1,
                          _Float16* __restrict__ Hh, int n) {
  int i = blockIdx.x * blockDim.x + threadIdx.x;
  int s = gridDim.x * blockDim.x;
  for (; i < n; i += s) {
    float v = agg[i] + b1[i & (INF_ - 1)];
    Hh[i] = (_Float16)(v > 0.f ? v : 0.f);
  }
}

// ---------------- layer 2 --------------------------------------------------

__global__ void gat_att2(const float* __restrict__ H2, const float* __restrict__ as,
                         const float* __restrict__ ad, float* __restrict__ a_s,
                         float* __restrict__ a_d, int N) {
  int n = blockIdx.x * blockDim.x + threadIdx.x;
  if (n >= N) return;
  const float* h = H2 + n * OUTF;
  float s = 0.f, d = 0.f;
#pragma unroll
  for (int c = 0; c < OUTF; ++c) {
    float v = h[c];
    s += v * as[c];
    d += v * ad[c];
  }
  a_s[n] = s; a_d[n] = d;
}

__global__ void gat_edge2(const int* __restrict__ src, const int* __restrict__ dst,
                          const float* __restrict__ a_s, const float* __restrict__ a_d,
                          float* __restrict__ ew, float* __restrict__ denom, int E) {
  int e = blockIdx.x * blockDim.x + threadIdx.x;
  if (e >= E) return;
  int s = src[e], d = dst[e];
  float v = a_s[s] + a_d[d];
  v = v > 0.f ? v : NEG_SLOPE * v;
  float w = expf(v);
  ew[e] = w;
  atomicAdd(&denom[d], w);
}

// wave-per-edge: lane l covers channels 2l, 2l+1
__global__ __launch_bounds__(256)
void gat_agg2(const int* __restrict__ src, const int* __restrict__ dst,
              const float* __restrict__ H2, const float* __restrict__ ew,
              const float* __restrict__ denom, float* __restrict__ out, int E) {
  int g = blockIdx.x * blockDim.x + threadIdx.x;
  int e = g >> 5, l = g & 31;
  if (e >= E) return;
  int s = src[e], d = dst[e];
  float coeff = ew[e] / (denom[d] + 1e-16f);
  const float2 hv = *reinterpret_cast<const float2*>(H2 + s * OUTF + l * 2);
  float* o = out + d * OUTF + l * 2;
  atomicAdd(o + 0, coeff * hv.x);
  atomicAdd(o + 1, coeff * hv.y);
}

__global__ void gat_bias2(float* __restrict__ out, const float* __restrict__ b2, int n) {
  int i = blockIdx.x * blockDim.x + threadIdx.x;
  int s = gridDim.x * blockDim.x;
  for (; i < n; i += s) out[i] += b2[i & (OUTF - 1)];
}

// ---------------- launcher -------------------------------------------------

extern "C" void kernel_launch(void* const* d_in, const int* in_sizes, int n_in,
                              void* d_out, int out_size, void* d_ws, size_t ws_size,
                              hipStream_t stream) {
  const float* x    = (const float*)d_in[0];
  const int*   esrc = (const int*)d_in[1];
  const int*   edst = (const int*)d_in[2];
  const float* W1   = (const float*)d_in[3];
  const float* as1  = (const float*)d_in[4];
  const float* ad1  = (const float*)d_in[5];
  const float* b1   = (const float*)d_in[6];
  const float* W2   = (const float*)d_in[7];
  const float* as2  = (const float*)d_in[8];
  const float* ad2  = (const float*)d_in[9];
  const float* b2   = (const float*)d_in[10];
  float* out = (float*)d_out;

  const int N = in_sizes[0] / INF_;
  const int E = in_sizes[1];

  char* ws = (char*)d_ws;
  size_t off = 0;
  auto alloc = [&](size_t bytes) -> void* {
    void* p = ws + off;
    off = (off + bytes + 255) & ~(size_t)255;
    return p;
  };

  _Float16*     Xh   = (_Float16*)    alloc((size_t)N * INF_ * 2);
  unsigned int* W1p  = (unsigned int*)alloc((size_t)64 * INF_ * 4);  // K/2 x 128
  unsigned int* W2p  = (unsigned int*)alloc((size_t)64 * OUTF * 4);  // K/2 x 64
  float*        H1   = (float*)       alloc((size_t)N * INF_ * 4);
  float*        aS1  = (float*)       alloc((size_t)N * HEADS * 4);
  float*        aD1  = (float*)       alloc((size_t)N * HEADS * 4);
  float*        ew1  = (float*)       alloc((size_t)E * HEADS * 4);
  float*        dn1  = (float*)       alloc((size_t)N * HEADS * 4);
  float*        agg1 = (float*)       alloc((size_t)N * INF_ * 4);
  _Float16*     Hh   = (_Float16*)    alloc((size_t)N * INF_ * 2);
  float*        H2   = (float*)       alloc((size_t)N * OUTF * 4);
  float*        aS2  = (float*)       alloc((size_t)N * 4);
  float*        aD2  = (float*)       alloc((size_t)N * 4);
  float*        ew2  = (float*)       alloc((size_t)E * 4);
  float*        dn2  = (float*)       alloc((size_t)N * 4);

  const int T = 256;
  auto blk = [](long n, int t) { return (unsigned)((n + t - 1) / t); };

  // init accumulators (d_out/d_ws are poisoned; re-zero every call)
  gat_zero_f32<<<blk((long)N * INF_, T), T, 0, stream>>>(agg1, N * INF_);
  gat_zero_f32<<<blk((long)N * HEADS, T), T, 0, stream>>>(dn1, N * HEADS);
  gat_zero_f32<<<blk((long)N, T), T, 0, stream>>>(dn2, N);
  gat_zero_f32<<<blk((long)N * OUTF, T), T, 0, stream>>>(out, N * OUTF);

  // precision conversion + WMMA B-layout packing
  gat_cvt_f16<<<blk((long)N * INF_, T), T, 0, stream>>>(x, Xh, N * INF_);
  gat_pack_w<<<blk(64 * INF_, T), T, 0, stream>>>(W1, W1p, 64, INF_);
  gat_pack_w<<<blk(64 * OUTF, T), T, 0, stream>>>(W2, W2p, 64, OUTF);

  // layer 1: H1 = Xh @ W1  (wave per 16x128 strip: 8 tiles, 32 WMMAs)
  {
    int waves = N / 16;
    gat_gemm_wmma<8><<<blk(waves, 4), 128, 0, stream>>>(
        (const unsigned int*)Xh, W1p, H1, N);
  }
  gat_att1<<<blk((long)N * HEADS, T), T, 0, stream>>>(H1, as1, ad1, aS1, aD1, N);
  gat_edge1<<<blk((long)E, T), T, 0, stream>>>(esrc, edst, aS1, aD1, ew1, dn1, E);
  gat_agg1<<<blk((long)E * 32, T), T, 0, stream>>>(esrc, edst, H1, ew1, dn1, agg1, E);
  gat_post1<<<blk((long)N * INF_, T), T, 0, stream>>>(agg1, b1, Hh, N * INF_);

  // layer 2: H2 = relu(agg1+b1) @ W2  (wave per 16x64 strip: 4 tiles, 16 WMMAs)
  {
    int waves = N / 16;
    gat_gemm_wmma<4><<<blk(waves, 4), 128, 0, stream>>>(
        (const unsigned int*)Hh, W2p, H2, N);
  }
  gat_att2<<<blk((long)N, T), T, 0, stream>>>(H2, as2, ad2, aS2, aD2, N);
  gat_edge2<<<blk((long)E, T), T, 0, stream>>>(esrc, edst, aS2, aD2, ew2, dn2, E);
  gat_agg2<<<blk((long)E * 32, T), T, 0, stream>>>(esrc, edst, H2, ew2, dn2, out, E);
  gat_bias2<<<blk((long)N * OUTF, T), T, 0, stream>>>(out, b2, N * OUTF);
}
